// FlaxLLaMAAttention_35046933135698
// MI455X (gfx1250) — matrix operand order repaired and checked by reading, not compiled, not run on hardware
//
#include <hip/hip_runtime.h>
#include <hip/hip_bf16.h>

// ---------------------------------------------------------------------------
// Problem constants (FlaxLLaMAAttention reference): B=2, S=2048, D=2048, H=16
// ---------------------------------------------------------------------------
#define BB 2
#define SS 2048
#define DD 2048
#define HH 16
#define HDIM 128
#define MM (BB * SS)          // 4096 token rows
#define QT_PER_BH (SS / 16)   // 128 query tiles per (b,h)

typedef __bf16 v16bf __attribute__((ext_vector_type(16)));
typedef __bf16 v8bf  __attribute__((ext_vector_type(8)));
typedef float  v8f   __attribute__((ext_vector_type(8)));

union V16 { v16bf v; v8bf h[2]; };

static __device__ __forceinline__ v8f wmma_bf16(v16bf a, v16bf b, v8f c) {
    return __builtin_amdgcn_wmma_f32_16x16x32_bf16(false, a, false, b,
                                                   (short)0, c, false, false);
}

// ---------------------------------------------------------------------------
// 1) f32 -> bf16 copy (activations)
// ---------------------------------------------------------------------------
__global__ void cvt_bf16_kernel(const float* __restrict__ in,
                                __bf16* __restrict__ out, int n) {
    int i = blockIdx.x * blockDim.x + threadIdx.x;
    if (i < n) out[i] = (__bf16)in[i];
}

// ---------------------------------------------------------------------------
// 2) W[k,n] f32 -> Wt[n,k] bf16 (LDS-tiled transpose)
// ---------------------------------------------------------------------------
__global__ void transpose_bf16_kernel(const float* __restrict__ in,
                                      __bf16* __restrict__ out) {
    __shared__ float tile[16][17];
    int k0 = blockIdx.x * 16, n0 = blockIdx.y * 16;
    tile[threadIdx.y][threadIdx.x] =
        in[(size_t)(k0 + threadIdx.y) * DD + n0 + threadIdx.x];
    __syncthreads();
    out[(size_t)(n0 + threadIdx.y) * DD + k0 + threadIdx.x] =
        (__bf16)tile[threadIdx.x][threadIdx.y];
}

// ---------------------------------------------------------------------------
// 3) bf16 WMMA GEMM: C[M=4096, N=2048] = A[M,K=2048] @ Bt[N,K]^T
//    One wave computes a 16x64 strip (4 accumulator tiles) so each A fragment
//    feeds 4 WMMAs.  4 waves/block -> block covers 16x256 of C.
//    MODE 0: C -> f32 row-major.  MODE 1: C -> bf16 V-transposed [B,H,HD,S].
// ---------------------------------------------------------------------------
template <int MODE>
__global__ __launch_bounds__(128, 1)
void gemm16_kernel(const __bf16* __restrict__ A,
                   const __bf16* __restrict__ Bt,
                   float* __restrict__ Cf,
                   __bf16* __restrict__ Cv) {
    const int K = DD, N = DD;
    int wave = threadIdx.x >> 5;
    int lane = threadIdx.x & 31;
    int g = lane >> 4;        // half-wave group
    int l16 = lane & 15;
    int nBase = (blockIdx.x * 4 + wave) * 64;   // 64 columns per wave
    int mBase = blockIdx.y * 16;

    const __bf16* aRow = A + (size_t)(mBase + l16) * K;   // A row for this lane
    const __bf16* bRow0 = Bt + (size_t)(nBase + l16) * K; // Wt rows = W columns
    const __bf16* bRow1 = bRow0 + (size_t)16 * K;
    const __bf16* bRow2 = bRow0 + (size_t)32 * K;
    const __bf16* bRow3 = bRow0 + (size_t)48 * K;

    v8f acc[4];
#pragma unroll
    for (int j = 0; j < 4; ++j) acc[j] = (v8f){};

    for (int k = 0; k < K; k += 32) {
        V16 af;
        // A in contiguous k order (same as B): lane holds row M=l16, K 16g..16g+15 contig
        af.h[0] = *(const v8bf*)(aRow + k + 16 * g);
        af.h[1] = *(const v8bf*)(aRow + k + 16 * g + 8);
        const __bf16* br[4] = {bRow0, bRow1, bRow2, bRow3};
#pragma unroll
        for (int j = 0; j < 4; ++j) {
            // B in contiguous k order (same as A): lane holds col N=l16, K 16g..16g+15 contig
            V16 bf;
            bf.h[0] = *(const v8bf*)(br[j] + k + 16 * g);
            bf.h[1] = *(const v8bf*)(br[j] + k + 16 * g + 8);
            acc[j] = wmma_bf16(af.v, bf.v, acc[j]);
        }
    }

    if (MODE == 0) {
#pragma unroll
        for (int j = 0; j < 4; ++j) {
#pragma unroll
            for (int r = 0; r < 8; ++r) {
                int m = mBase + r + 8 * g;
                Cf[(size_t)m * N + nBase + 16 * j + l16] = acc[j][r];
            }
        }
    } else {
        // V output: row m=(b,s), col n=(h,d) -> Vt[((b*H+h)*HD+d)*S + s] bf16
#pragma unroll
        for (int j = 0; j < 4; ++j) {
            int n = nBase + 16 * j + l16;
            int h = n >> 7, d = n & (HDIM - 1);
#pragma unroll
            for (int r = 0; r < 8; ++r) {
                int m = mBase + r + 8 * g;
                int b = m >> 11, s = m & (SS - 1);
                Cv[((size_t)(b * HH + h) * HDIM + d) * SS + s] = (__bf16)acc[j][r];
            }
        }
    }
}

// ---------------------------------------------------------------------------
// 4) RoPE + repack: Qf/Kf [B,S,D] f32 -> Qbf/Kbf [B,H,S,HD] bf16
// ---------------------------------------------------------------------------
__global__ void rope_pack_kernel(const float* __restrict__ Qf,
                                 const float* __restrict__ Kf,
                                 const int* __restrict__ pos_ids,
                                 __bf16* __restrict__ Qbf,
                                 __bf16* __restrict__ Kbf) {
    const size_t half = (size_t)BB * SS * HH * (HDIM / 2);
    size_t id = (size_t)blockIdx.x * blockDim.x + threadIdx.x;
    bool isK = id >= half;
    size_t t = isK ? id - half : id;
    int j = (int)(t % (HDIM / 2)); t /= (HDIM / 2);
    int h = (int)(t % HH);         t /= HH;
    int s = (int)(t % SS);
    int b = (int)(t / SS);

    const float* src = isK ? Kf : Qf;
    __bf16* dst = isK ? Kbf : Qbf;

    size_t inIdx = ((size_t)(b * SS + s)) * DD + h * HDIM + 2 * j;
    float a = src[inIdx], bv = src[inIdx + 1];
    float pos = (float)pos_ids[b * SS + s];
    // inv_freq = theta^(-2j/HD) ; ln(1e4) = 9.21034
    float freq = __expf((-2.0f * (float)j / (float)HDIM) * 9.210340372f);
    float ang = pos * freq;
    float sn, cs;
    __sincosf(ang, &sn, &cs);
    float ra = a * cs - bv * sn;
    float rb = a * sn + bv * cs;

    size_t outIdx = (((size_t)(b * HH + h) * SS + s)) * HDIM + 2 * j;
    dst[outIdx] = (__bf16)ra;
    dst[outIdx + 1] = (__bf16)rb;
}

// ---------------------------------------------------------------------------
// 5) Flash attention: one wave per 16-query tile of one (b,h).
//    Online softmax; 32 keys per step: 8 score WMMAs + 8 P@V WMMAs.
//    P is transposed to the WMMA A-layout through LDS (1KB per wave).
// ---------------------------------------------------------------------------
__global__ __launch_bounds__(128, 1)
void flash_attn_kernel(const __bf16* __restrict__ Qbf,
                       const __bf16* __restrict__ Kbf,
                       const __bf16* __restrict__ Vt,
                       const int* __restrict__ amask,
                       __bf16* __restrict__ Obf) {
    __shared__ __align__(16) __bf16 lds[4][16][32];
    int wave = threadIdx.x >> 5;
    int lane = threadIdx.x & 31;
    int g = lane >> 4, l16 = lane & 15;

    int tile = blockIdx.x * 4 + wave;        // 0 .. B*H*(S/16)-1 = 4095
    int qt = tile & (QT_PER_BH - 1);
    int bh = tile >> 7;                      // 0..31
    int b = bh >> 4;

    __bf16(*P)[32] = lds[wave];

    const __bf16* Qh = Qbf + (size_t)bh * SS * HDIM;
    const __bf16* Kh = Kbf + (size_t)bh * SS * HDIM;
    const __bf16* Vh = Vt + (size_t)bh * HDIM * SS;

    // Resident Q fragments: 4 chunks of K-dim (HD=128 -> 4 x 32)
    V16 qa[4];
    const __bf16* qRow = Qh + (size_t)(qt * 16 + l16) * HDIM;
#pragma unroll
    for (int c = 0; c < 4; ++c) {
        qa[c].h[0] = *(const v8bf*)(qRow + 32 * c + 16 * g);
        qa[c].h[1] = *(const v8bf*)(qRow + 32 * c + 16 * g + 8);
    }

    v8f out[8];
#pragma unroll
    for (int t = 0; t < 8; ++t) out[t] = (v8f){};
    float mrow[8], lrow[8];
#pragma unroll
    for (int r = 0; r < 8; ++r) { mrow[r] = -3.0e38f; lrow[r] = 0.0f; }

    const float scale = 0.08838834764831845f;   // 1/sqrt(128)
    const float NEG = -3.0e38f;
    int nkb = (qt * 16 + 16 + 31) / 32;         // ceil((#valid keys)/32)

    for (int kb = 0; kb < nkb * 32; kb += 32) {
        // --- scores: two 16x16 tiles over this 32-key block -----------------
        v8f sc[2];
        sc[0] = (v8f){}; sc[1] = (v8f){};
#pragma unroll
        for (int t = 0; t < 2; ++t) {
            const __bf16* kRow = Kh + (size_t)(kb + 16 * t + l16) * HDIM;
#pragma unroll
            for (int c = 0; c < 4; ++c) {
                V16 kf;
                kf.h[0] = *(const v8bf*)(kRow + 32 * c + 16 * g);
                kf.h[1] = *(const v8bf*)(kRow + 32 * c + 16 * g + 8);
                sc[t] = wmma_bf16(qa[c].v, kf.v, sc[t]);
            }
        }
        // --- scale + causal + padding mask ---------------------------------
#pragma unroll
        for (int t = 0; t < 2; ++t) {
            int k = kb + 16 * t + l16;
            float padd = (amask[b * SS + k] != 0) ? 0.0f : NEG;
#pragma unroll
            for (int r = 0; r < 8; ++r) {
                int sq = qt * 16 + r + 8 * g;
                float v = sc[t][r] * scale + padd;
                sc[t][r] = (k <= sq) ? v : NEG;
            }
        }
        // --- online softmax (row stats live per 16-lane group) -------------
        float corr[8];
#pragma unroll
        for (int r = 0; r < 8; ++r) {
            float mx = fmaxf(sc[0][r], sc[1][r]);
#pragma unroll
            for (int off = 8; off >= 1; off >>= 1)
                mx = fmaxf(mx, __shfl_xor(mx, off, 32));
            float mnew = fmaxf(mrow[r], mx);
            corr[r] = __expf(mrow[r] - mnew);
            mrow[r] = mnew;
            float p0 = __expf(sc[0][r] - mnew);
            float p1 = __expf(sc[1][r] - mnew);
            sc[0][r] = p0; sc[1][r] = p1;
            float ps = p0 + p1;
#pragma unroll
            for (int off = 8; off >= 1; off >>= 1)
                ps += __shfl_xor(ps, off, 32);
            lrow[r] = lrow[r] * corr[r] + ps;
        }
#pragma unroll
        for (int t = 0; t < 8; ++t)
#pragma unroll
            for (int r = 0; r < 8; ++r) out[t][r] *= corr[r];

        // --- transpose P (C-layout -> A-layout) through LDS ----------------
#pragma unroll
        for (int t = 0; t < 2; ++t)
#pragma unroll
            for (int r = 0; r < 8; ++r)
                P[r + 8 * g][16 * t + l16] = (__bf16)sc[t][r];
        asm volatile("s_wait_dscnt 0" ::: "memory");
        V16 pa;
        pa.h[0] = *(const v8bf*)(&P[l16][16 * g]);
        pa.h[1] = *(const v8bf*)(&P[l16][16 * g + 8]);

        // --- P @ V : 8 output d-tiles --------------------------------------
#pragma unroll
        for (int t = 0; t < 8; ++t) {
            const __bf16* vRow = Vh + (size_t)(t * 16 + l16) * SS + kb;
            V16 vf;
            vf.h[0] = *(const v8bf*)(vRow + 16 * g);
            vf.h[1] = *(const v8bf*)(vRow + 16 * g + 8);
            out[t] = wmma_bf16(pa.v, vf.v, out[t]);
        }
    }

    // --- epilogue: normalize and store bf16 in [B,S,D] (h-major) -----------
    int h = bh & (HH - 1);
#pragma unroll
    for (int r = 0; r < 8; ++r) {
        float inv = 1.0f / lrow[r];
        int sq = qt * 16 + r + 8 * g;
        size_t base = ((size_t)(b * SS + sq)) * DD + h * HDIM;
#pragma unroll
        for (int t = 0; t < 8; ++t)
            Obf[base + t * 16 + l16] = (__bf16)(out[t][r] * inv);
    }
}

// ---------------------------------------------------------------------------
// Launcher
// ---------------------------------------------------------------------------
extern "C" void kernel_launch(void* const* d_in, const int* in_sizes, int n_in,
                              void* d_out, int out_size, void* d_ws, size_t ws_size,
                              hipStream_t stream) {
    const float* X = (const float*)d_in[0];
    const int* amask = (const int*)d_in[1];
    const int* pos = (const int*)d_in[2];
    const float* wq = (const float*)d_in[3];
    const float* wk = (const float*)d_in[4];
    const float* wv = (const float*)d_in[5];
    const float* wo = (const float*)d_in[6];
    float* out = (float*)d_out;

    char* w = (char*)d_ws;
    __bf16* Xbf = (__bf16*)w;            w += (size_t)MM * DD * 2;
    __bf16* WqT = (__bf16*)w;            w += (size_t)DD * DD * 2;
    __bf16* WkT = (__bf16*)w;            w += (size_t)DD * DD * 2;
    __bf16* WvT = (__bf16*)w;            w += (size_t)DD * DD * 2;
    __bf16* WoT = (__bf16*)w;            w += (size_t)DD * DD * 2;
    float* Qf   = (float*)w;             w += (size_t)MM * DD * 4;
    float* Kf   = (float*)w;             w += (size_t)MM * DD * 4;
    __bf16* Qbf = (__bf16*)w;            w += (size_t)MM * DD * 2;
    __bf16* Kbf = (__bf16*)w;            w += (size_t)MM * DD * 2;
    __bf16* Vtb = (__bf16*)w;            w += (size_t)MM * DD * 2;
    __bf16* Obf = (__bf16*)Qf;           // alias: Qf dead after rope_pack

    // 1) activations to bf16
    {
        int n = MM * DD;
        cvt_bf16_kernel<<<(n + 255) / 256, 256, 0, stream>>>(X, Xbf, n);
    }
    // 2) weight transposes to bf16
    {
        dim3 g(DD / 16, DD / 16), blk(16, 16);
        transpose_bf16_kernel<<<g, blk, 0, stream>>>(wq, WqT);
        transpose_bf16_kernel<<<g, blk, 0, stream>>>(wk, WkT);
        transpose_bf16_kernel<<<g, blk, 0, stream>>>(wv, WvT);
        transpose_bf16_kernel<<<g, blk, 0, stream>>>(wo, WoT);
    }
    // 3) QKV projections (V stored directly transposed bf16)
    {
        dim3 g(DD / 256, MM / 16);   // 4 waves/block, 64 cols/wave
        gemm16_kernel<0><<<g, 128, 0, stream>>>(Xbf, WqT, Qf, (__bf16*)nullptr);
        gemm16_kernel<0><<<g, 128, 0, stream>>>(Xbf, WkT, Kf, (__bf16*)nullptr);
        gemm16_kernel<1><<<g, 128, 0, stream>>>(Xbf, WvT, (float*)nullptr, Vtb);
    }
    // 4) RoPE + repack Q/K
    {
        size_t n = (size_t)2 * BB * SS * HH * (HDIM / 2);   // 8.39M
        rope_pack_kernel<<<(unsigned)((n + 255) / 256), 256, 0, stream>>>(
            Qf, Kf, pos, Qbf, Kbf);
    }
    // 5) flash attention (4096 wave-tiles, 4 waves/block)
    flash_attn_kernel<<<(BB * HH * QT_PER_BH) / 4, 128, 0, stream>>>(
        Qbf, Kbf, Vtb, amask, Obf);
    // 6) output projection -> f32 d_out
    {
        dim3 g(DD / 256, MM / 16);
        gemm16_kernel<0><<<g, 128, 0, stream>>>(Obf, WoT, out, (__bf16*)nullptr);
    }
}
